// GatedCrossAttentionFusion_25185688224185
// MI455X (gfx1250) — matrix operand order repaired and checked by reading, not compiled, not run on hardware
//
#include <hip/hip_runtime.h>
#include <hip/hip_bf16.h>
#include <math.h>

// ---------------------------------------------------------------------------
// Problem constants (from reference)
// ---------------------------------------------------------------------------
#define B_SZ   16384
#define VD_SZ  2048
#define SD_SZ  1024
#define CD_SZ  512
#define H_SZ   256
#define NHEADS 4
#define DH_SZ  64
#define ROWS3  (B_SZ * 3)       // 49152 rows for the stacked tensors
#define LN_EPS 1e-5f

typedef __bf16 bf16_t;
typedef bf16_t v16bf __attribute__((ext_vector_type(16)));
typedef bf16_t v8bf  __attribute__((ext_vector_type(8)));
typedef bf16_t v4bf  __attribute__((ext_vector_type(4)));
typedef float  v8f   __attribute__((ext_vector_type(8)));
typedef float  v4f   __attribute__((ext_vector_type(4)));

// ---------------------------------------------------------------------------
// WMMA-bf16 GEMM tile kernel:  Y[row, wcol+col] = X[row,:K] @ W[wcol+col,:K]^T + bias
// Block = 256 threads (8 waves). Block computes a 32-row x 256-col slab.
// Wave w owns col tiles 2w, 2w+1 across both 16-row tiles -> 4 accumulators.
// Software-pipelined: global loads for chunk c+1 are issued before the WMMAs
// of chunk c, converted+stored to the alternate LDS buffer afterwards
// (double-buffered LDS, one barrier per K-chunk).
// DO_LN: fused per-row LayerNorm over the 256 output cols (requires grid.y==1).
// residual (optional, DO_LN path): added before LN, indexed with out_stride.
// ---------------------------------------------------------------------------
template <bool DO_LN>
__global__ __launch_bounds__(256) void gemm_tile_kernel(
    const float* __restrict__ X,        // [M, K]
    const float* __restrict__ W,        // [>= wcol+256, K]
    const float* __restrict__ bias,     // [>= wcol+256]
    const float* __restrict__ residual, // may be null; [M, out_stride]
    const float* __restrict__ gamma,    // [256] (DO_LN)
    const float* __restrict__ beta,     // [256] (DO_LN)
    float* __restrict__ Y,              // base (caller pre-offsets), row stride out_stride
    int K, int out_stride)
{
    __shared__ __align__(64) bf16_t sA[2][32][32];    // 2x 2 KB   activation tiles
    __shared__ __align__(64) bf16_t sB[2][256][32];   // 2x 16 KB  weight tiles
    __shared__ __align__(64) float  sOut[32][256];    // 32 KB     epilogue staging

    const int tid  = threadIdx.x;
    const int wave = tid >> 5;
    const int lane = tid & 31;
    const int half = lane >> 4;      // A-frag K-half / C-frag M-half selector
    const int mrow = lane & 15;      // A-frag M row within a 16-row tile
    const int ncol = lane & 15;      // B/C-frag N col
    const unsigned row0 = blockIdx.x << 5;          // 32 rows per block
    const unsigned wcol = blockIdx.y << 8;

    const int t0 = wave * 2;
    const int t1 = wave * 2 + 1;

    // float4 staging mapping: 8 threads cover one 32-float row
    const int ldr = tid >> 3;          // 0..31
    const int ldc = (tid & 7) * 4;     // 0,4,...,28

    const float* Xp = X + (unsigned)(row0 + ldr) * K + ldc;
    const float* Wp = W + (unsigned)(wcol + ldr) * K + ldc;

    v8f acc00 = {}, acc01 = {}, acc10 = {}, acc11 = {};

    v4f areg;
    v4f wreg[8];

    // issue global loads for chunk starting at k0 (registers only)
    auto load_chunk = [&](int k0) {
        areg = *(const v4f*)(Xp + k0);
#pragma unroll
        for (int i = 0; i < 8; ++i)
            wreg[i] = *(const v4f*)(Wp + (unsigned)(32 * i) * K + k0);
    };
    // convert fp32 regs -> bf16 and commit to LDS buffer `buf`
    auto store_chunk = [&](int buf) {
        v4bf pa = { (bf16_t)areg[0], (bf16_t)areg[1], (bf16_t)areg[2], (bf16_t)areg[3] };
        *(v4bf*)&sA[buf][ldr][ldc] = pa;
#pragma unroll
        for (int i = 0; i < 8; ++i) {
            v4bf p = { (bf16_t)wreg[i][0], (bf16_t)wreg[i][1],
                       (bf16_t)wreg[i][2], (bf16_t)wreg[i][3] };
            *(v4bf*)&sB[buf][ldr + 32 * i][ldc] = p;
        }
    };

    // prologue: fill buffer 0
    load_chunk(0);
    store_chunk(0);
    __syncthreads();

    const int nchunks = K >> 5;
    for (int c = 0; c < nchunks; ++c) {
        const int cur = c & 1;
        const bool has_next = (c + 1) < nchunks;

        // start next chunk's global loads early (overlap with WMMA)
        if (has_next) load_chunk((c + 1) << 5);

        // ---- A fragments per CDNA5 16-bit 16x32 layout ----------------------
        v16bf afrag0, afrag1;
        {
            v8bf lo0 = *(const v8bf*)&sA[cur][mrow][half * 8];
            v8bf hi0 = *(const v8bf*)&sA[cur][mrow][16 + half * 8];
            v8bf lo1 = *(const v8bf*)&sA[cur][16 + mrow][half * 8];
            v8bf hi1 = *(const v8bf*)&sA[cur][16 + mrow][16 + half * 8];
#pragma unroll
            for (int i = 0; i < 8; ++i) {
                afrag0[i] = lo0[i]; afrag0[8 + i] = hi0[i];
                afrag1[i] = lo1[i]; afrag1[8 + i] = hi1[i];
            }
        }
        // ---- B fragments: same K order as A (K=8h..8h+7, 16+8h..16+8h+7) -----
        v16bf b0, b1;
        {
            v8bf blo0 = *(const v8bf*)&sB[cur][t0 * 16 + ncol][half * 8];
            v8bf bhi0 = *(const v8bf*)&sB[cur][t0 * 16 + ncol][16 + half * 8];
            v8bf blo1 = *(const v8bf*)&sB[cur][t1 * 16 + ncol][half * 8];
            v8bf bhi1 = *(const v8bf*)&sB[cur][t1 * 16 + ncol][16 + half * 8];
#pragma unroll
            for (int i = 0; i < 8; ++i) {
                b0[i] = blo0[i]; b0[8 + i] = bhi0[i];
                b1[i] = blo1[i]; b1[8 + i] = bhi1[i];
            }
        }

        acc00 = __builtin_amdgcn_wmma_f32_16x16x32_bf16(false, afrag0, false, b0,
                                                        (short)0, acc00, false, false);
        acc01 = __builtin_amdgcn_wmma_f32_16x16x32_bf16(false, afrag0, false, b1,
                                                        (short)0, acc01, false, false);
        acc10 = __builtin_amdgcn_wmma_f32_16x16x32_bf16(false, afrag1, false, b0,
                                                        (short)0, acc10, false, false);
        acc11 = __builtin_amdgcn_wmma_f32_16x16x32_bf16(false, afrag1, false, b1,
                                                        (short)0, acc11, false, false);

        if (has_next) {
            store_chunk(cur ^ 1);   // waits on loadcnt here, behind the WMMAs
            __syncthreads();
        }
    }

    // ---- epilogue: C layout VGPR r -> M = r + 8*half, N = ncol --------------
    const int c0 = t0 * 16 + ncol;
    const int c1 = t1 * 16 + ncol;
    const float bia0 = bias[wcol + c0];
    const float bia1 = bias[wcol + c1];
#pragma unroll
    for (int r = 0; r < 8; ++r) {
        int m0 = r + 8 * half;       // row tile 0
        int m1 = 16 + m0;            // row tile 1
        float v00 = acc00[r] + bia0;
        float v01 = acc01[r] + bia1;
        float v10 = acc10[r] + bia0;
        float v11 = acc11[r] + bia1;
        if (residual != nullptr) {
            v00 += residual[(unsigned)(row0 + m0) * out_stride + wcol + c0];
            v01 += residual[(unsigned)(row0 + m0) * out_stride + wcol + c1];
            v10 += residual[(unsigned)(row0 + m1) * out_stride + wcol + c0];
            v11 += residual[(unsigned)(row0 + m1) * out_stride + wcol + c1];
        }
        if (DO_LN) {
            sOut[m0][c0] = v00; sOut[m0][c1] = v01;
            sOut[m1][c0] = v10; sOut[m1][c1] = v11;
        } else {
            Y[(unsigned)(row0 + m0) * out_stride + wcol + c0] = v00;
            Y[(unsigned)(row0 + m0) * out_stride + wcol + c1] = v01;
            Y[(unsigned)(row0 + m1) * out_stride + wcol + c0] = v10;
            Y[(unsigned)(row0 + m1) * out_stride + wcol + c1] = v11;
        }
    }

    if (DO_LN) {
        __syncthreads();
        // wave w normalizes rows 4w..4w+3; stride-32 column access (bank-clean)
#pragma unroll
        for (int rr = 0; rr < 4; ++rr) {
            int m = wave * 4 + rr;
            float s = 0.f, sq = 0.f;
#pragma unroll
            for (int i = 0; i < 8; ++i) {
                float v = sOut[m][lane + 32 * i];
                s += v; sq += v * v;
            }
#pragma unroll
            for (int off = 16; off >= 1; off >>= 1) {
                s  += __shfl_xor(s,  off, 32);
                sq += __shfl_xor(sq, off, 32);
            }
            float mean = s * (1.f / 256.f);
            float var  = sq * (1.f / 256.f) - mean * mean;
            float rstd = rsqrtf(var + LN_EPS);
#pragma unroll
            for (int i = 0; i < 8; ++i) {
                int c = lane + 32 * i;
                float v = (sOut[m][c] - mean) * rstd * gamma[c] + beta[c];
                Y[(unsigned)(row0 + m) * out_stride + c] = v;   // coalesced 128B
            }
        }
    }
}

// ---------------------------------------------------------------------------
// Tiny attention: one thread per (batch, head). seq-len 3, head dim 64.
// qkv: [B*3, 768] rows; ctx out: [B*3, 256]
// ---------------------------------------------------------------------------
__global__ __launch_bounds__(256) void attn_kernel(
    const float* __restrict__ qkv, float* __restrict__ ctx)
{
    int t = blockIdx.x * blockDim.x + threadIdx.x;
    if (t >= B_SZ * NHEADS) return;
    int b = t >> 2;
    int h = t & 3;
    const float* base = qkv + (unsigned)b * 3 * 768 + h * DH_SZ;

    float sc[3][3] = {{0.f}};
    for (int d = 0; d < DH_SZ; d += 4) {
        v4f q[3], k[3];
#pragma unroll
        for (int i = 0; i < 3; ++i) {
            q[i] = *(const v4f*)(base + i * 768 + d);
            k[i] = *(const v4f*)(base + i * 768 + 256 + d);
        }
#pragma unroll
        for (int i = 0; i < 3; ++i)
#pragma unroll
            for (int j = 0; j < 3; ++j)
#pragma unroll
                for (int e = 0; e < 4; ++e)
                    sc[i][j] += q[i][e] * k[j][e];
    }

    float attn[3][3];
    const float scale = 0.125f; // 1/sqrt(64)
#pragma unroll
    for (int i = 0; i < 3; ++i) {
        float m = fmaxf(sc[i][0], fmaxf(sc[i][1], sc[i][2])) * scale;
        float e0 = __expf(sc[i][0] * scale - m);
        float e1 = __expf(sc[i][1] * scale - m);
        float e2 = __expf(sc[i][2] * scale - m);
        float inv = 1.f / (e0 + e1 + e2);
        attn[i][0] = e0 * inv; attn[i][1] = e1 * inv; attn[i][2] = e2 * inv;
    }

    for (int d = 0; d < DH_SZ; d += 4) {
        v4f v0 = *(const v4f*)(base + 0 * 768 + 512 + d);
        v4f v1 = *(const v4f*)(base + 1 * 768 + 512 + d);
        v4f v2 = *(const v4f*)(base + 2 * 768 + 512 + d);
#pragma unroll
        for (int i = 0; i < 3; ++i) {
            v4f o;
#pragma unroll
            for (int e = 0; e < 4; ++e)
                o[e] = attn[i][0] * v0[e] + attn[i][1] * v1[e] + attn[i][2] * v2[e];
            *(v4f*)(ctx + (unsigned)(b * 3 + i) * H_SZ + h * DH_SZ + d) = o;
        }
    }
}

// ---------------------------------------------------------------------------
// Gating MLP + fuse. One 64-thread block per batch row.
// attended: [B, 768] (== [B*3, 256] grouped by 3 -> concat(av,as,ac) contiguous)
// ---------------------------------------------------------------------------
__global__ __launch_bounds__(64) void gate_kernel(
    const float* __restrict__ attended,
    const float* __restrict__ g1w, const float* __restrict__ g1b,
    const float* __restrict__ g2w, const float* __restrict__ g2b,
    float* __restrict__ fused, float* __restrict__ gates)
{
    __shared__ float x[768];
    __shared__ float h1[64];
    __shared__ float lg[3];

    const int b = blockIdx.x;
    const int j = threadIdx.x;
    const float* xr = attended + (unsigned)b * 768;

#pragma unroll
    for (int i = 0; i < 3; ++i)
        *(v4f*)&x[(j + 64 * i) * 4] = *(const v4f*)&xr[(j + 64 * i) * 4];
    __syncthreads();

    // h1[j] = gelu_exact( g1w[j,:] . x + g1b[j] )
    float acc = g1b[j];
    const float* wr = g1w + (unsigned)j * 768;
#pragma unroll 4
    for (int k = 0; k < 768; k += 4) {
        v4f w4 = *(const v4f*)&wr[k];
        acc += w4[0] * x[k] + w4[1] * x[k + 1] + w4[2] * x[k + 2] + w4[3] * x[k + 3];
    }
    h1[j] = 0.5f * acc * (1.f + erff(acc * 0.70710678118654752f));
    __syncthreads();

    if (j < 3) {
        float l = g2b[j];
        const float* w2 = g2w + j * 64;
#pragma unroll 8
        for (int k = 0; k < 64; ++k) l += w2[k] * h1[k];
        lg[j] = l;
    }
    __syncthreads();

    float m  = fmaxf(lg[0], fmaxf(lg[1], lg[2]));
    float e0 = __expf(lg[0] - m), e1 = __expf(lg[1] - m), e2 = __expf(lg[2] - m);
    float inv = 1.f / (e0 + e1 + e2);
    float g0 = e0 * inv, g1v = e1 * inv, g2v = e2 * inv;

    const int c = j * 4;
    v4f av = *(const v4f*)(x + c);
    v4f as = *(const v4f*)(x + 256 + c);
    v4f ac = *(const v4f*)(x + 512 + c);
    v4f o;
#pragma unroll
    for (int e = 0; e < 4; ++e)
        o[e] = g0 * av[e] + g1v * as[e] + g2v * ac[e];
    *(v4f*)(fused + (unsigned)b * H_SZ + c) = o;

    if (j < 3) gates[(unsigned)b * 3 + j] = (j == 0) ? g0 : ((j == 1) ? g1v : g2v);
}

// ---------------------------------------------------------------------------
// Launch
// ---------------------------------------------------------------------------
extern "C" void kernel_launch(void* const* d_in, const int* in_sizes, int n_in,
                              void* d_out, int out_size, void* d_ws, size_t ws_size,
                              hipStream_t stream)
{
    const float* fv   = (const float*)d_in[0];
    const float* fs   = (const float*)d_in[1];
    const float* fc   = (const float*)d_in[2];
    const float* Wv   = (const float*)d_in[3];
    const float* bv   = (const float*)d_in[4];
    const float* gv   = (const float*)d_in[5];
    const float* bev  = (const float*)d_in[6];
    const float* Ws   = (const float*)d_in[7];
    const float* bs   = (const float*)d_in[8];
    const float* gs   = (const float*)d_in[9];
    const float* bes  = (const float*)d_in[10];
    const float* Wc   = (const float*)d_in[11];
    const float* bc   = (const float*)d_in[12];
    const float* gc_  = (const float*)d_in[13];
    const float* bec  = (const float*)d_in[14];
    const float* in_w = (const float*)d_in[15];
    const float* in_b = (const float*)d_in[16];
    const float* out_w= (const float*)d_in[17];
    const float* out_b= (const float*)d_in[18];
    const float* gn   = (const float*)d_in[19];
    const float* gb   = (const float*)d_in[20];
    const float* g1w  = (const float*)d_in[21];
    const float* g1b  = (const float*)d_in[22];
    const float* g2w  = (const float*)d_in[23];
    const float* g2b  = (const float*)d_in[24];

    // workspace layout (fp32)
    float* stack    = (float*)d_ws;                       // [49152, 256]
    float* qkv      = stack + (size_t)ROWS3 * H_SZ;       // [49152, 768]
    float* ctx      = qkv   + (size_t)ROWS3 * 3 * H_SZ;   // [49152, 256]
    float* attended = ctx   + (size_t)ROWS3 * H_SZ;       // [49152, 256] == [B, 768]

    float* fused = (float*)d_out;                         // [B, 256]
    float* gates = fused + (size_t)B_SZ * H_SZ;           // [B, 3]

    const dim3 blk(256);

    // Stage 1: three branch projections + LayerNorm -> stack[b][branch][:]
    gemm_tile_kernel<true><<<dim3(B_SZ / 32, 1), blk, 0, stream>>>(
        fv, Wv, bv, nullptr, gv, bev, stack + 0 * H_SZ, VD_SZ, 3 * H_SZ);
    gemm_tile_kernel<true><<<dim3(B_SZ / 32, 1), blk, 0, stream>>>(
        fs, Ws, bs, nullptr, gs, bes, stack + 1 * H_SZ, SD_SZ, 3 * H_SZ);
    gemm_tile_kernel<true><<<dim3(B_SZ / 32, 1), blk, 0, stream>>>(
        fc, Wc, bc, nullptr, gc_, bec, stack + 2 * H_SZ, CD_SZ, 3 * H_SZ);

    // Stage 2: qkv = stack @ in_w^T + in_b   (N=768 split across grid.y)
    gemm_tile_kernel<false><<<dim3(ROWS3 / 32, 3), blk, 0, stream>>>(
        stack, in_w, in_b, nullptr, nullptr, nullptr, qkv, H_SZ, 3 * H_SZ);

    // Stage 3a: tiny attention -> ctx
    attn_kernel<<<dim3((B_SZ * NHEADS) / 256), blk, 0, stream>>>(qkv, ctx);

    // Stage 3b: attended = LN(ctx @ out_w^T + out_b + stack)
    gemm_tile_kernel<true><<<dim3(ROWS3 / 32, 1), blk, 0, stream>>>(
        ctx, out_w, out_b, stack, gn, gb, attended, H_SZ, H_SZ);

    // Stage 4: gating MLP + fuse
    gate_kernel<<<dim3(B_SZ), dim3(64), 0, stream>>>(
        attended, g1w, g1b, g2w, g2b, fused, gates);
}
